// LinearAttention_10333691314265
// MI455X (gfx1250) — compile-verified, matrix operand order checked by reading, not run on hardware
//
#include <hip/hip_runtime.h>
#include <stdint.h>

#define L_SEQ 2048
#define DIM   1024
#define NH    16
#define DH    64
#define CHUNK 64
#define NC    (L_SEQ / CHUNK)   // 32
#define EPS_F 1e-6f

typedef __attribute__((ext_vector_type(16))) __bf16 bf16x16;
typedef __attribute__((ext_vector_type(8)))  float  v8f;

union FragU { uint32_t u[8]; bf16x16 v; };

__device__ __forceinline__ v8f v8f_zero() {
    v8f z;
#pragma unroll
    for (int i = 0; i < 8; i++) z[i] = 0.0f;
    return z;
}

__device__ __forceinline__ uint16_t f2bf(float x) {
    union { float f; uint32_t u; } c; c.f = x;
    uint32_t u = c.u;
    uint32_t r = (u + 0x7FFFu + ((u >> 16) & 1u)) >> 16;   // round-nearest-even
    return (uint16_t)r;
}
__device__ __forceinline__ float bf2f(uint16_t b) {
    union { uint32_t u; float f; } c; c.u = ((uint32_t)b) << 16;
    return c.f;
}

// ---- CDNA5 async global->LDS DMA (ASYNCcnt-tracked) ----
// VDST = VGPR with LDS byte offset, VADDR = 64-bit global address.
// INST_OFFSET is added to BOTH the LDS and global addresses (ISA 15.18.3),
// so a 64B row copy shares one LDS VGPR + one vaddr pair across 4 b128 ops.
__device__ __forceinline__ uint32_t lds_off(const void* p) {
    return (uint32_t)(uintptr_t)p;   // generic LDS pointer: low 32 bits = LDS offset
}
__device__ __forceinline__ void async_row64(uint32_t lds, const void* g) {
    asm volatile(
        "global_load_async_to_lds_b128 %0, %1, off\n\t"
        "global_load_async_to_lds_b128 %0, %1, off offset:16\n\t"
        "global_load_async_to_lds_b128 %0, %1, off offset:32\n\t"
        "global_load_async_to_lds_b128 %0, %1, off offset:48"
        :: "v"(lds), "v"(g) : "memory");
}
#define WAIT_ASYNC0() asm volatile("s_wait_asynccnt 0x0" ::: "memory")

// Load one 16x32 bf16 A/B fragment from LDS (row-major [row][k], stride in u16, even).
// CDNA5 16-bit matrix layout: lane<16 -> K pairs {0,1},{2,3},{4,5},{6,7},
// {16,17},{18,19},{20,21},{22,23}; lanes 16-31 add +8 to every K.
__device__ __forceinline__ bf16x16 load_frag(const uint16_t* base, int row, int stride,
                                             int lane, int k0) {
    const int half = (lane >> 4) & 1;
    const uint16_t* rp = base + row * stride + k0 + half * 8;
    FragU f;
    f.u[0] = *(const uint32_t*)(rp + 0);
    f.u[1] = *(const uint32_t*)(rp + 2);
    f.u[2] = *(const uint32_t*)(rp + 4);
    f.u[3] = *(const uint32_t*)(rp + 6);
    f.u[4] = *(const uint32_t*)(rp + 16);
    f.u[5] = *(const uint32_t*)(rp + 18);
    f.u[6] = *(const uint32_t*)(rp + 20);
    f.u[7] = *(const uint32_t*)(rp + 22);
    return f.v;
}

__device__ __forceinline__ v8f wmma_bf16(bf16x16 a, bf16x16 b, v8f c) {
    return __builtin_amdgcn_wmma_f32_16x16x32_bf16(false, a, false, b, (short)0, c,
                                                   false, false);
}

// ---------------------------------------------------------------- converts
__global__ __launch_bounds__(256) void cvt_bf16_kernel(const float* __restrict__ in,
                                                       uint16_t* __restrict__ out, int n) {
    int i = blockIdx.x * 256 + threadIdx.x;
    if (i < n) out[i] = f2bf(in[i]);
}

// out[n][k] = bf16(w[k][n])  (pre-transpose weights for the B operand)
__global__ __launch_bounds__(256) void cvt_wT_kernel(const float* __restrict__ w,
                                                     uint16_t* __restrict__ wT, int dim) {
    int i = blockIdx.x * 256 + threadIdx.x;
    if (i < dim * dim) {
        int n = i / dim, k = i % dim;
        wT[i] = f2bf(w[(size_t)k * dim + n]);
    }
}

// ---------------------------------------------------------------- GEMM (bf16 -> f32)
// C[M][N] = A[M][K] * B[K][N], B supplied transposed: BT[N][K].
// Tile 128x128x64, double-buffered async-to-LDS staging.
#define GT_M 128
#define GT_N 128
#define GT_K 64
#define GSTR 80   // u16 stride; 160B rows -> 16B aligned

__global__ __launch_bounds__(256)
void gemm_bf16_kernel(const uint16_t* __restrict__ A, const uint16_t* __restrict__ BT,
                      float* __restrict__ C, int M, int N, int K) {
    __shared__ uint16_t As[2][GT_M * GSTR];
    __shared__ uint16_t Bs[2][GT_N * GSTR];
    const int t = threadIdx.x, lane = t & 31, wid = t >> 5;
    const int wm = wid & 3;       // 4 row strips of 32
    const int wn = wid >> 2;      // 2 col strips of 64
    const int m0 = blockIdx.y * GT_M;
    const int n0 = blockIdx.x * GT_N;

    v8f acc[2][4];
#pragma unroll
    for (int i = 0; i < 2; i++)
#pragma unroll
        for (int j = 0; j < 4; j++) acc[i][j] = v8f_zero();

    const int lrow  = t >> 1;            // 0..127
    const int lcolu = (t & 1) * 32;      // u16 offset 0 / 32

    const uint16_t* gA = A  + (size_t)(m0 + lrow) * K + lcolu;
    const uint16_t* gB = BT + (size_t)(n0 + lrow) * K + lcolu;
    const uint32_t  lA0 = lds_off(&As[0][lrow * GSTR + lcolu]);
    const uint32_t  lB0 = lds_off(&Bs[0][lrow * GSTR + lcolu]);
    const uint32_t  bufStep = (uint32_t)(GT_M * GSTR * 2);   // bytes between buffers

    const int NS = K / GT_K;
    // prologue: stage 0
    async_row64(lA0, gA);
    async_row64(lB0, gB);

    for (int s = 0; s < NS; s++) {
        WAIT_ASYNC0();
        __syncthreads();
        if (s + 1 < NS) {
            uint32_t boff = (uint32_t)((s + 1) & 1) * bufStep;
            async_row64(lA0 + boff, gA + (size_t)(s + 1) * GT_K);
            async_row64(lB0 + boff, gB + (size_t)(s + 1) * GT_K);
        }
        const uint16_t* Ab = As[s & 1];
        const uint16_t* Bb = Bs[s & 1];
#pragma unroll
        for (int kk = 0; kk < GT_K; kk += 32) {
            bf16x16 af[2], bfr[4];
#pragma unroll
            for (int mt = 0; mt < 2; mt++)
                af[mt] = load_frag(Ab, wm * 32 + mt * 16 + (lane & 15), GSTR, lane, kk);
#pragma unroll
            for (int nt = 0; nt < 4; nt++)
                bfr[nt] = load_frag(Bb, wn * 64 + nt * 16 + (lane & 15), GSTR, lane, kk);
#pragma unroll
            for (int mt = 0; mt < 2; mt++)
#pragma unroll
                for (int nt = 0; nt < 4; nt++)
                    acc[mt][nt] = wmma_bf16(af[mt], bfr[nt], acc[mt][nt]);
        }
    }

    // epilogue: one 64-bit base per lane, constant column offsets (imm-offset stores)
    const int half = lane >> 4, nn = lane & 15;
    float* cbase = C + (size_t)(m0 + wm * 32 + half * 8) * N + (n0 + wn * 64 + nn);
#pragma unroll
    for (int mt = 0; mt < 2; mt++)
#pragma unroll
        for (int g = 0; g < 8; g++) {
            float* rp = cbase + (size_t)(mt * 16 + g) * N;
#pragma unroll
            for (int nt = 0; nt < 4; nt++) rp[nt * 16] = acc[mt][nt][g];
        }
}

// ---------------------------------------------------------------- RoPE + ReLU + pack
// Inputs qf/kf/vf are f32 [L][DIM]. Outputs (bf16):
//   q_bf,k_bf : [H][L][DH]   (A operand / B operand of QK^T)
//   kT_bf,vT_bf: [H][DH][L]  (A of K^T V, B^T of PV)
__global__ __launch_bounds__(256)
void rope_pack_kernel(const float* __restrict__ qf, const float* __restrict__ kf,
                      const float* __restrict__ vf, uint16_t* __restrict__ q_bf,
                      uint16_t* __restrict__ k_bf, uint16_t* __restrict__ kT_bf,
                      uint16_t* __restrict__ vT_bf) {
    int idx = blockIdx.x * 256 + threadIdx.x;
    if (idx >= L_SEQ * NH) return;
    const int h = idx & (NH - 1);
    const int l = idx >> 4;
    const float* qr = qf + (size_t)l * DIM + h * DH;
    const float* kr = kf + (size_t)l * DIM + h * DH;
    const float* vr = vf + (size_t)l * DIM + h * DH;
    uint16_t* qo = q_bf + ((size_t)h * L_SEQ + l) * DH;
    uint16_t* ko = k_bf + ((size_t)h * L_SEQ + l) * DH;
    const float fl = (float)l;
#pragma unroll 4
    for (int p = 0; p < DH / 2; p++) {
        float expo = (float)(2 * p) / (float)DH;
        float inv  = __powf(10000.0f, -expo);
        float ang  = fl * inv;
        float s, c;
        __sincosf(ang, &s, &c);
        float q1 = qr[2 * p], q2 = qr[2 * p + 1];
        float k1 = kr[2 * p], k2 = kr[2 * p + 1];
        float rq0 = fmaxf(q1 * c - q2 * s, 0.0f);
        float rq1 = fmaxf(q1 * s + q2 * c, 0.0f);
        float rk0 = fmaxf(k1 * c - k2 * s, 0.0f);
        float rk1 = fmaxf(k1 * s + k2 * c, 0.0f);
        qo[2 * p] = f2bf(rq0); qo[2 * p + 1] = f2bf(rq1);
        ko[2 * p] = f2bf(rk0); ko[2 * p + 1] = f2bf(rk1);
        kT_bf[((size_t)h * DH + 2 * p) * L_SEQ + l]     = f2bf(rk0);
        kT_bf[((size_t)h * DH + 2 * p + 1) * L_SEQ + l] = f2bf(rk1);
        vT_bf[((size_t)h * DH + 2 * p) * L_SEQ + l]     = f2bf(vr[2 * p]);
        vT_bf[((size_t)h * DH + 2 * p + 1) * L_SEQ + l] = f2bf(vr[2 * p + 1]);
    }
}

// ---------------------------------------------------------------- per-chunk KV = K^T V
#define CSTR 80   // u16 stride; 160B rows -> 16B aligned

__global__ __launch_bounds__(128)
void chunk_kv_kernel(const uint16_t* __restrict__ kT_bf, const uint16_t* __restrict__ vT_bf,
                     const uint16_t* __restrict__ k_bf, float* __restrict__ kvT,
                     float* __restrict__ ksum) {
    __shared__ uint16_t kTs[CHUNK * CSTR];
    __shared__ uint16_t vTs[CHUNK * CSTR];
    const int t = threadIdx.x, lane = t & 31, w = t >> 5;
    const int h = blockIdx.x >> 5;
    const int c = blockIdx.x & (NC - 1);
    {
        int row = t >> 1, col0 = (t & 1) * 32;
        async_row64(lds_off(&kTs[row * CSTR + col0]),
                    kT_bf + ((size_t)h * DH + row) * L_SEQ + c * CHUNK + col0);
        async_row64(lds_off(&vTs[row * CSTR + col0]),
                    vT_bf + ((size_t)h * DH + row) * L_SEQ + c * CHUNK + col0);
    }
    WAIT_ASYNC0();
    __syncthreads();

    v8f acc[4];
#pragma unroll
    for (int nt = 0; nt < 4; nt++) acc[nt] = v8f_zero();
    const int d0 = w * 16;
#pragma unroll
    for (int ks0 = 0; ks0 < CHUNK; ks0 += 32) {
        bf16x16 a = load_frag(kTs, d0 + (lane & 15), CSTR, lane, ks0);
#pragma unroll
        for (int nt = 0; nt < 4; nt++) {
            bf16x16 b = load_frag(vTs, nt * 16 + (lane & 15), CSTR, lane, ks0);
            acc[nt] = wmma_bf16(a, b, acc[nt]);
        }
    }
    // store KV transposed: kvT[h][c][m][d]  (so the scanned state is S^T, B^T of Q@S)
    const int half = lane >> 4, nn = lane & 15;
    float* out = kvT + (size_t)(h * NC + c) * DH * DH;
#pragma unroll
    for (int nt = 0; nt < 4; nt++)
#pragma unroll
        for (int g = 0; g < 8; g++) {
            int d = d0 + g + half * 8;
            int m = nt * 16 + nn;
            out[m * DH + d] = acc[nt][g];
        }
    if (t < DH) {
        float s = 0.0f;
        const uint16_t* kb = k_bf + ((size_t)h * L_SEQ + c * CHUNK) * DH + t;
        for (int l = 0; l < CHUNK; l++) s += bf2f(kb[l * DH]);
        ksum[(size_t)(h * NC + c) * DH + t] = s;
    }
}

// ---------------------------------------------------------------- exclusive prefix scan over chunks
__global__ __launch_bounds__(256)
void scan_kernel(const float* __restrict__ kvT, const float* __restrict__ ksum,
                 uint16_t* __restrict__ s_bf, float* __restrict__ ksum_pref) {
    const int h = blockIdx.x, t = threadIdx.x;
    float acc[16];
#pragma unroll
    for (int j = 0; j < 16; j++) acc[j] = 0.0f;
    for (int c = 0; c < NC; c++) {
        const float* in = kvT + (size_t)(h * NC + c) * (DH * DH);
        uint16_t* out = s_bf + (size_t)(h * NC + c) * (DH * DH);
#pragma unroll
        for (int j = 0; j < 16; j++) {
            int e = t + j * 256;
            out[e] = f2bf(acc[j]);
            acc[j] += in[e];
        }
    }
    if (t < DH) {
        float a = 0.0f;
        for (int c = 0; c < NC; c++) {
            ksum_pref[(size_t)(h * NC + c) * DH + t] = a;
            a += ksum[(size_t)(h * NC + c) * DH + t];
        }
    }
}

// ---------------------------------------------------------------- per-chunk output
// P = causal(Q K^T); num = P V + Q S_prev; den = rowsum(P) + Q . ksum_prev
__global__ __launch_bounds__(128)
void chunk_out_kernel(const uint16_t* __restrict__ q_bf, const uint16_t* __restrict__ k_bf,
                      const uint16_t* __restrict__ vT_bf, const uint16_t* __restrict__ s_bf,
                      const float* __restrict__ ksum_pref, uint16_t* __restrict__ attn_bf) {
    __shared__ uint16_t qs[CHUNK * CSTR];
    __shared__ uint16_t ks[CHUNK * CSTR];
    __shared__ uint16_t vTs[CHUNK * CSTR];
    __shared__ uint16_t ss[CHUNK * CSTR];
    __shared__ uint16_t Ps[CHUNK * CSTR];
    __shared__ float denom[CHUNK];
    const int t = threadIdx.x, lane = t & 31, w = t >> 5;
    const int h = blockIdx.x >> 5;
    const int c = blockIdx.x & (NC - 1);
    {
        int row = t >> 1, col0 = (t & 1) * 32;
        async_row64(lds_off(&qs[row * CSTR + col0]),
                    q_bf + ((size_t)h * L_SEQ + c * CHUNK + row) * DH + col0);
        async_row64(lds_off(&ks[row * CSTR + col0]),
                    k_bf + ((size_t)h * L_SEQ + c * CHUNK + row) * DH + col0);
        async_row64(lds_off(&vTs[row * CSTR + col0]),
                    vT_bf + ((size_t)h * DH + row) * L_SEQ + c * CHUNK + col0);
        async_row64(lds_off(&ss[row * CSTR + col0]),
                    s_bf + (size_t)(h * NC + c) * (DH * DH) + row * DH + col0);
    }
    WAIT_ASYNC0();
    __syncthreads();

    const int half = lane >> 4, nn = lane & 15;
    const int i0 = w * 16;

    // phase 1: P = Q K^T, causal mask, rowsums, stash bf16 P in LDS
    v8f p[4];
#pragma unroll
    for (int nt = 0; nt < 4; nt++) p[nt] = v8f_zero();
#pragma unroll
    for (int ks0 = 0; ks0 < CHUNK; ks0 += 32) {
        bf16x16 a = load_frag(qs, i0 + nn, CSTR, lane, ks0);
#pragma unroll
        for (int nt = 0; nt < 4; nt++) {
            bf16x16 b = load_frag(ks, nt * 16 + nn, CSTR, lane, ks0);
            p[nt] = wmma_bf16(a, b, p[nt]);
        }
    }
    float rs[8];
#pragma unroll
    for (int g = 0; g < 8; g++) rs[g] = 0.0f;
#pragma unroll
    for (int nt = 0; nt < 4; nt++)
#pragma unroll
        for (int g = 0; g < 8; g++) {
            int i = i0 + g + half * 8;
            int j = nt * 16 + nn;
            float val = (j <= i) ? p[nt][g] : 0.0f;
            rs[g] += val;
            Ps[i * CSTR + j] = f2bf(val);
        }
#pragma unroll
    for (int g = 0; g < 8; g++) {
        float r = rs[g];
        r += __shfl_xor(r, 8, 16);
        r += __shfl_xor(r, 4, 16);
        r += __shfl_xor(r, 2, 16);
        r += __shfl_xor(r, 1, 16);
        if (nn == 0) denom[i0 + g + half * 8] = r;
    }
    __syncthreads();

    // phase 2: add Q . ksum_prefix, invert
    if (t < CHUNK) {
        float dsum = denom[t];
        const float* kp = ksum_pref + (size_t)(h * NC + c) * DH;
        for (int d = 0; d < DH; d++) dsum += bf2f(qs[t * CSTR + d]) * kp[d];
        denom[t] = 1.0f / (dsum + EPS_F);
    }
    __syncthreads();

    // phase 3: num = Q S^T' + P V
    v8f num[4];
#pragma unroll
    for (int nt = 0; nt < 4; nt++) num[nt] = v8f_zero();
#pragma unroll
    for (int ks0 = 0; ks0 < CHUNK; ks0 += 32) {
        bf16x16 aq = load_frag(qs, i0 + nn, CSTR, lane, ks0);
        bf16x16 ap = load_frag(Ps, i0 + nn, CSTR, lane, ks0);
#pragma unroll
        for (int nt = 0; nt < 4; nt++) {
            bf16x16 bs = load_frag(ss, nt * 16 + nn, CSTR, lane, ks0);
            num[nt] = wmma_bf16(aq, bs, num[nt]);
            bf16x16 bv = load_frag(vTs, nt * 16 + nn, CSTR, lane, ks0);
            num[nt] = wmma_bf16(ap, bv, num[nt]);
        }
    }
#pragma unroll
    for (int nt = 0; nt < 4; nt++)
#pragma unroll
        for (int g = 0; g < 8; g++) {
            int i = i0 + g + half * 8;
            int m = nt * 16 + nn;
            float val = num[nt][g] * denom[i];
            attn_bf[((size_t)(c * CHUNK + i)) * DIM + h * DH + m] = f2bf(val);
        }
}

// ---------------------------------------------------------------- host
extern "C" void kernel_launch(void* const* d_in, const int* in_sizes, int n_in,
                              void* d_out, int out_size, void* d_ws, size_t ws_size,
                              hipStream_t stream) {
    (void)in_sizes; (void)n_in; (void)out_size; (void)ws_size;
    const float* x  = (const float*)d_in[0];
    const float* wq = (const float*)d_in[1];
    const float* wk = (const float*)d_in[2];
    const float* wv = (const float*)d_in[3];
    const float* wo = (const float*)d_in[4];
    float* out = (float*)d_out;

    char* p = (char*)d_ws;
    auto alloc = [&](size_t b) -> void* {
        void* r = (void*)p;
        p += (b + 255) & ~(size_t)255;
        return r;
    };
    uint16_t* xb   = (uint16_t*)alloc((size_t)L_SEQ * DIM * 2);
    uint16_t* wqT  = (uint16_t*)alloc((size_t)DIM * DIM * 2);
    uint16_t* wkT  = (uint16_t*)alloc((size_t)DIM * DIM * 2);
    uint16_t* wvT  = (uint16_t*)alloc((size_t)DIM * DIM * 2);
    uint16_t* woT  = (uint16_t*)alloc((size_t)DIM * DIM * 2);
    float*    qf   = (float*)alloc((size_t)L_SEQ * DIM * 4);
    float*    kf   = (float*)alloc((size_t)L_SEQ * DIM * 4);
    float*    vf   = (float*)alloc((size_t)L_SEQ * DIM * 4);
    uint16_t* q_bf = (uint16_t*)alloc((size_t)L_SEQ * DIM * 2);
    uint16_t* k_bf = (uint16_t*)alloc((size_t)L_SEQ * DIM * 2);
    uint16_t* kT_bf = (uint16_t*)alloc((size_t)L_SEQ * DIM * 2);
    uint16_t* vT_bf = (uint16_t*)alloc((size_t)L_SEQ * DIM * 2);
    float*    kvT  = (float*)alloc((size_t)NH * NC * DH * DH * 4);
    float*    ksum = (float*)alloc((size_t)NH * NC * DH * 4);
    float*    ksum_pref = (float*)alloc((size_t)NH * NC * DH * 4);
    uint16_t* s_bf = (uint16_t*)alloc((size_t)NH * NC * DH * DH * 2);
    uint16_t* attn_bf = (uint16_t*)alloc((size_t)L_SEQ * DIM * 2);

    const int nX = L_SEQ * DIM;
    cvt_bf16_kernel<<<(nX + 255) / 256, 256, 0, stream>>>(x, xb, nX);
    const int nW = DIM * DIM;
    cvt_wT_kernel<<<(nW + 255) / 256, 256, 0, stream>>>(wq, wqT, DIM);
    cvt_wT_kernel<<<(nW + 255) / 256, 256, 0, stream>>>(wk, wkT, DIM);
    cvt_wT_kernel<<<(nW + 255) / 256, 256, 0, stream>>>(wv, wvT, DIM);
    cvt_wT_kernel<<<(nW + 255) / 256, 256, 0, stream>>>(wo, woT, DIM);

    dim3 gg(DIM / GT_N, L_SEQ / GT_M);
    gemm_bf16_kernel<<<gg, 256, 0, stream>>>(xb, wqT, qf, L_SEQ, DIM, DIM);
    gemm_bf16_kernel<<<gg, 256, 0, stream>>>(xb, wkT, kf, L_SEQ, DIM, DIM);
    gemm_bf16_kernel<<<gg, 256, 0, stream>>>(xb, wvT, vf, L_SEQ, DIM, DIM);

    rope_pack_kernel<<<(L_SEQ * NH + 255) / 256, 256, 0, stream>>>(qf, kf, vf, q_bf, k_bf,
                                                                   kT_bf, vT_bf);
    chunk_kv_kernel<<<NH * NC, 128, 0, stream>>>(kT_bf, vT_bf, k_bf, kvT, ksum);
    scan_kernel<<<NH, 256, 0, stream>>>(kvT, ksum, s_bf, ksum_pref);
    chunk_out_kernel<<<NH * NC, 128, 0, stream>>>(q_bf, k_bf, vT_bf, s_bf, ksum_pref,
                                                  attn_bf);
    gemm_bf16_kernel<<<gg, 256, 0, stream>>>(attn_bf, woT, out, L_SEQ, DIM, DIM);
}
